// TopkMultiscaleGNN_49246095016473
// MI455X (gfx1250) — compile-verified, matrix operand order checked by reading, not run on hardware
//
#include <hip/hip_runtime.h>
#include <hip/hip_bf16.h>

// MI455X / gfx1250, wave32, WMMA bf16 16x16x32 with f32 accumulate.
// Problem is HBM-bound on the edge feature array (~1.2 GB total traffic,
// ~53us at 23.3 TB/s); x/agg/weights are L2-resident (192MB L2).
// Weights are copied into LDS with GLOBAL_LOAD_ASYNC_TO_LDS_B128 (ASYNCcnt).

#define CCH 64           // channels
#define LAYERS 3

typedef __attribute__((ext_vector_type(16))) __bf16 v16bf;
typedef __attribute__((ext_vector_type(8)))  float  v8f;

// ---------------------------------------------------------------------------
// CDNA5 async global->LDS copy (CDNA5 ISA):
//   LDS address is taken from the VDST VGPR; tracked with ASYNCcnt.
// LDS address = low 32 bits of the generic pointer (flat LDS aperture rule).
// ---------------------------------------------------------------------------
__device__ __forceinline__ void async_copy_b128(unsigned lds_addr, const void* gaddr) {
    asm volatile("global_load_async_to_lds_b128 %0, %1, off"
                 :: "v"(lds_addr), "v"(gaddr) : "memory");
}
__device__ __forceinline__ void wait_async_all() {
    asm volatile("s_wait_asynccnt 0x0" ::: "memory");
}

// ---------------------------------------------------------------------------
// Fragment layout helpers (CDNA5 ISA 7.12.2, 16-bit A 16x32):
//   lane = m + 16*half ; per-lane j=0..15 covers k = (j&7) | (half<<3) | ((j&8)<<1)
// B fragments mirrored: lane = n + 16*half with the same k(j) mapping.
// A fragment for K-step ks lives at frag[ks*512 + lane*16 + j] (bf16).
// ---------------------------------------------------------------------------

__global__ __launch_bounds__(32)
void swizzle_weight_kernel(const float* __restrict__ W, __bf16* __restrict__ out) {
    // one block = one (kstep, ntile) B-fragment; W is [Ktot x 64] row-major f32
    const int frag  = blockIdx.x;
    const int kstep = frag >> 2;
    const int nt    = frag & 3;
    const int lane  = threadIdx.x;
    const int n     = nt * 16 + (lane & 15);
    const int half  = lane >> 4;
#pragma unroll
    for (int j = 0; j < 16; ++j) {
        int k = kstep * 32 + ((j & 7) | (half << 3) | ((j & 8) << 1));
        out[frag * 512 + lane * 16 + j] = (__bf16)W[k * 64 + n];
    }
}

// D = A(16xK) * B(Kx64) accumulated over `ksteps` K-slices of 32
__device__ __forceinline__
void gemm_frags(v8f* acc, const __bf16* afr, const __bf16* bfr, int ksteps, int lane) {
    for (int ks = 0; ks < ksteps; ++ks) {
        v16bf a = *(const v16bf*)(afr + ks * 512 + lane * 16);
#pragma unroll
        for (int nt = 0; nt < 4; ++nt) {
            v16bf b = *(const v16bf*)(bfr + (ks * 4 + nt) * 512 + lane * 16);
            acc[nt] = __builtin_amdgcn_wmma_f32_16x16x32_bf16(
                false, a, false, b, (short)0, acc[nt], false, false);
        }
    }
}

// C-fragment (16x16 f32, 8 VGPRs: m = r + 8*(lane>>4), n = nt*16 + (lane&15))
// -> relu(c + bias[n]) stored to LDS in A-fragment layout (channel n becomes K)
__device__ __forceinline__
void cfrag_relu_to_afrag(const v8f* acc, const float* __restrict__ bias,
                         __bf16* hb, int lane) {
#pragma unroll
    for (int nt = 0; nt < 4; ++nt) {
        int n   = nt * 16 + (lane & 15);
        float bb = bias[n];
        int kk   = n & 31;
        int half = (kk >> 3) & 1;
        int j    = (kk & 7) | ((kk & 16) >> 1);
        __bf16* dst = hb + (n >> 5) * 512 + j;
#pragma unroll
        for (int r = 0; r < 8; ++r) {
            int m = r + ((lane >> 4) << 3);
            float h = acc[nt][r] + bb;
            h = h > 0.f ? h : 0.f;
            dst[(m + (half << 4)) * 16] = (__bf16)h;
        }
    }
}

// ---------------------------------------------------------------------------
// Edge kernel: e += LN(MLP(cat(x[src], x[dst], e))) ; agg[dst] += e (atomic)
// 128 threads = 4 waves; each wave owns a 16-edge tile.
// Dynamic LDS: [0,40KB) weight frags (w0 24KB | w1 8KB | w2 8KB),
//              [40KB + wave*6KB): staging (6KB) overlaid by hbuf(2KB)+lnbuf(4KB)
// ---------------------------------------------------------------------------
__global__ __launch_bounds__(128)
void edge_mlp_kernel(const float* __restrict__ x, float* __restrict__ e,
                     const int* __restrict__ srcI, const int* __restrict__ dstI,
                     const __bf16* __restrict__ w0f, const __bf16* __restrict__ w1f,
                     const __bf16* __restrict__ w2f,
                     const float* __restrict__ b0, const float* __restrict__ b1,
                     const float* __restrict__ b2,
                     const float* __restrict__ g, const float* __restrict__ beta,
                     float* __restrict__ agg, int E) {
    extern __shared__ char smem[];
    __bf16* w0l = (__bf16*)smem;           // 12288 bf16
    __bf16* w1l = w0l + 12288;             // 4096 bf16
    __bf16* w2l = w0l + 16384;             // 4096 bf16
    const int tid  = threadIdx.x;
    const int lane = tid & 31;
    const int wave = tid >> 5;
    char*   my    = smem + 40 * 1024 + wave * (6 * 1024);
    __bf16* stg   = (__bf16*)my;           // 6 ksteps * 512 = 6KB (GEMM0 input)
    __bf16* hbuf  = (__bf16*)my;           // 2 ksteps * 512 = 2KB (overlay)
    float*  lnbuf = (float*)(my + 2 * 1024); // 16*64 f32 = 4KB (overlay)

    // async copy of pre-swizzled weight fragments -> LDS (ASYNCcnt path)
    {
        const unsigned lds0 = (unsigned)(size_t)(void*)smem;
        for (int i = tid; i < 1536; i += 128)                 // w0: 24KB
            async_copy_b128(lds0 + i * 16, (const char*)w0f + (size_t)i * 16);
        for (int i = tid; i < 512; i += 128)                  // w1: 8KB
            async_copy_b128(lds0 + 24576 + i * 16, (const char*)w1f + (size_t)i * 16);
        for (int i = tid; i < 512; i += 128)                  // w2: 8KB
            async_copy_b128(lds0 + 32768 + i * 16, (const char*)w2f + (size_t)i * 16);
    }

    const int eb = (blockIdx.x * 4 + wave) * 16;

    // prefetch next tile's e rows (the only HBM-streamed array)
    __builtin_prefetch(e + (size_t)(eb + 64) * CCH, 0, 1);

    // per-lane row indices (lanes 0..15 own rows 0..15)
    int myEdge = eb + (lane & 15);
    int ce     = myEdge < E ? myEdge : (E - 1);
    int sIdx   = srcI[ce];
    int dIdx   = dstI[ce];

    // stage cat(x[s], x[d], e) 16x192 f32 -> bf16 A-fragment layout in LDS
    for (int it = lane; it < 768; it += 32) {          // 768 float4 chunks
        int r    = it / 48;                            // row (edge) 0..15
        int c4   = it % 48;                            // float4 chunk in row
        int edge = eb + r;
        int s = __shfl(sIdx, r, 32);
        int d = __shfl(dIdx, r, 32);
        const float* base = (c4 < 16) ? (x + (size_t)s * CCH)
                          : (c4 < 32) ? (x + (size_t)d * CCH)
                                      : (e + (size_t)edge * CCH);
        float4 v;
        if (edge < E) v = *(const float4*)(base + (c4 & 15) * 4);
        else          v = make_float4(0.f, 0.f, 0.f, 0.f);
        float vv[4] = {v.x, v.y, v.z, v.w};
        int k0 = c4 * 4;
#pragma unroll
        for (int fi = 0; fi < 4; ++fi) {
            int k = k0 + fi, kk = k & 31;
            int half = (kk >> 3) & 1;
            int j    = (kk & 7) | ((kk & 16) >> 1);
            stg[(k >> 5) * 512 + (r + (half << 4)) * 16 + j] = (__bf16)vv[fi];
        }
    }
    wait_async_all();                       // weight frags resident in LDS
    __syncthreads();

    v8f acc[4] = {};                        // GEMM0: K=192 (6 ksteps)
    gemm_frags(acc, stg, w0l, 6, lane);
    __syncthreads();                        // stg region reused as hbuf
    cfrag_relu_to_afrag(acc, b0, hbuf, lane);
    __syncthreads();

    v8f acc2[4] = {};                       // GEMM1: K=64 (2 ksteps)
    gemm_frags(acc2, hbuf, w1l, 2, lane);
    __syncthreads();
    cfrag_relu_to_afrag(acc2, b1, hbuf, lane);
    __syncthreads();

    v8f acc3[4] = {};                       // GEMM2: K=64
    gemm_frags(acc3, hbuf, w2l, 2, lane);

    // h2 + b2 -> lnbuf (f32, row-major 16x64)
#pragma unroll
    for (int nt = 0; nt < 4; ++nt) {
        int n = nt * 16 + (lane & 15);
        float bb = b2[n];
#pragma unroll
        for (int r = 0; r < 8; ++r) {
            int m = r + ((lane >> 4) << 3);
            lnbuf[m * CCH + n] = acc3[nt][r] + bb;
        }
    }
    __syncthreads();

    // LayerNorm + residual + scatter-add: 2 lanes per row, 32 cols each
    {
        int row   = lane >> 1;
        int halfc = (lane & 1) * 32;
        const float* hr = lnbuf + row * CCH + halfc;
        float s1 = 0.f, s2 = 0.f;
#pragma unroll
        for (int i2 = 0; i2 < 32; ++i2) { float v = hr[i2]; s1 += v; s2 += v * v; }
        s1 += __shfl_xor(s1, 1, 32);
        s2 += __shfl_xor(s2, 1, 32);
        float mean = s1 * (1.f / 64.f);
        float var  = s2 * (1.f / 64.f) - mean * mean;
        float rs   = rsqrtf(var + 1e-5f);
        int edge = eb + row;
        int dn   = __shfl(dIdx, row, 32);
        if (edge < E) {
            float* erow = e   + (size_t)edge * CCH;
            float* arow = agg + (size_t)dn   * CCH;
#pragma unroll
            for (int i2 = 0; i2 < 32; ++i2) {
                int n = halfc + i2;
                float out  = (hr[i2] - mean) * rs * g[n] + beta[n];
                float enew = erow[n] + out;
                erow[n] = enew;
                unsafeAtomicAdd(arow + n, enew);   // global_atomic_add_f32 (L2)
            }
        }
    }
}

// ---------------------------------------------------------------------------
// Node kernel: x += LN(MLP(cat(x, agg)))
// LDS: [0,32KB) weights (w0 16KB | w1 8KB | w2 8KB), per-wave 6KB areas
// ---------------------------------------------------------------------------
__global__ __launch_bounds__(128)
void node_mlp_kernel(float* __restrict__ x, const float* __restrict__ agg,
                     const __bf16* __restrict__ w0f, const __bf16* __restrict__ w1f,
                     const __bf16* __restrict__ w2f,
                     const float* __restrict__ b0, const float* __restrict__ b1,
                     const float* __restrict__ b2,
                     const float* __restrict__ g, const float* __restrict__ beta,
                     int N) {
    extern __shared__ char smem[];
    __bf16* w0l = (__bf16*)smem;           // 8192 bf16
    __bf16* w1l = w0l + 8192;              // 4096 bf16
    __bf16* w2l = w0l + 12288;             // 4096 bf16
    const int tid  = threadIdx.x;
    const int lane = tid & 31;
    const int wave = tid >> 5;
    char*   my    = smem + 32 * 1024 + wave * (6 * 1024);
    __bf16* stg   = (__bf16*)my;           // 4 ksteps * 512 = 4KB
    __bf16* hbuf  = (__bf16*)my;           // overlay 2KB
    float*  lnbuf = (float*)(my + 2 * 1024); // overlay 4KB

    // async copy of weight fragments -> LDS
    {
        const unsigned lds0 = (unsigned)(size_t)(void*)smem;
        for (int i = tid; i < 1024; i += 128)                 // w0: 16KB
            async_copy_b128(lds0 + i * 16, (const char*)w0f + (size_t)i * 16);
        for (int i = tid; i < 512; i += 128)                  // w1: 8KB
            async_copy_b128(lds0 + 16384 + i * 16, (const char*)w1f + (size_t)i * 16);
        for (int i = tid; i < 512; i += 128)                  // w2: 8KB
            async_copy_b128(lds0 + 24576 + i * 16, (const char*)w2f + (size_t)i * 16);
    }

    const int nb = (blockIdx.x * 4 + wave) * 16;

    // stage cat(x[i], agg[i]) 16x128 -> A-fragment layout
    for (int it = lane; it < 512; it += 32) {          // 512 float4 chunks
        int r    = it / 32;
        int c4   = it % 32;
        int node = nb + r;
        const float* base = (c4 < 16) ? (x + (size_t)node * CCH)
                                      : (agg + (size_t)node * CCH);
        float4 v;
        if (node < N) v = *(const float4*)(base + (c4 & 15) * 4);
        else          v = make_float4(0.f, 0.f, 0.f, 0.f);
        float vv[4] = {v.x, v.y, v.z, v.w};
        int k0 = c4 * 4;
#pragma unroll
        for (int fi = 0; fi < 4; ++fi) {
            int k = k0 + fi, kk = k & 31;
            int half = (kk >> 3) & 1;
            int j    = (kk & 7) | ((kk & 16) >> 1);
            stg[(k >> 5) * 512 + (r + (half << 4)) * 16 + j] = (__bf16)vv[fi];
        }
    }
    wait_async_all();
    __syncthreads();

    v8f acc[4] = {};                        // GEMM0: K=128 (4 ksteps)
    gemm_frags(acc, stg, w0l, 4, lane);
    __syncthreads();
    cfrag_relu_to_afrag(acc, b0, hbuf, lane);
    __syncthreads();

    v8f acc2[4] = {};
    gemm_frags(acc2, hbuf, w1l, 2, lane);
    __syncthreads();
    cfrag_relu_to_afrag(acc2, b1, hbuf, lane);
    __syncthreads();

    v8f acc3[4] = {};
    gemm_frags(acc3, hbuf, w2l, 2, lane);

#pragma unroll
    for (int nt = 0; nt < 4; ++nt) {
        int n = nt * 16 + (lane & 15);
        float bb = b2[n];
#pragma unroll
        for (int r = 0; r < 8; ++r) {
            int m = r + ((lane >> 4) << 3);
            lnbuf[m * CCH + n] = acc3[nt][r] + bb;
        }
    }
    __syncthreads();

    {
        int row   = lane >> 1;
        int halfc = (lane & 1) * 32;
        const float* hr = lnbuf + row * CCH + halfc;
        float s1 = 0.f, s2 = 0.f;
#pragma unroll
        for (int i2 = 0; i2 < 32; ++i2) { float v = hr[i2]; s1 += v; s2 += v * v; }
        s1 += __shfl_xor(s1, 1, 32);
        s2 += __shfl_xor(s2, 1, 32);
        float mean = s1 * (1.f / 64.f);
        float var  = s2 * (1.f / 64.f) - mean * mean;
        float rs   = rsqrtf(var + 1e-5f);
        int node = nb + row;
        if (node < N) {
            float* xrow = x + (size_t)node * CCH;
#pragma unroll
            for (int i2 = 0; i2 < 32; ++i2) {
                int n = halfc + i2;
                float out = (hr[i2] - mean) * rs * g[n] + beta[n];
                xrow[n] = xrow[n] + out;
            }
        }
    }
}

// ---------------------------------------------------------------------------
extern "C" void kernel_launch(void* const* d_in, const int* in_sizes, int n_in,
                              void* d_out, int out_size, void* d_ws, size_t ws_size,
                              hipStream_t stream) {
    const int N = in_sizes[0] / CCH;
    const int E = in_sizes[1] / CCH;

    const float* x_in = (const float*)d_in[0];
    const float* e_in = (const float*)d_in[1];
    const int*   ei   = (const int*)d_in[2];
    const int*   srcI = ei;
    const int*   dstI = ei + E;
    const float* ew0 = (const float*)d_in[3];   const float* eb0 = (const float*)d_in[4];
    const float* ew1 = (const float*)d_in[5];   const float* eb1 = (const float*)d_in[6];
    const float* ew2 = (const float*)d_in[7];   const float* eb2 = (const float*)d_in[8];
    const float* eg  = (const float*)d_in[9];   const float* ebt = (const float*)d_in[10];
    const float* nw0 = (const float*)d_in[11];  const float* nb0 = (const float*)d_in[12];
    const float* nw1 = (const float*)d_in[13];  const float* nb1 = (const float*)d_in[14];
    const float* nw2 = (const float*)d_in[15];  const float* nb2 = (const float*)d_in[16];
    const float* ng  = (const float*)d_in[17];  const float* nbt = (const float*)d_in[18];

    // d_out = [x (N*C) | e (E*C)] f32 working buffers
    float* xbuf = (float*)d_out;
    float* ebuf = xbuf + (size_t)N * CCH;
    hipMemcpyAsync(xbuf, x_in, (size_t)N * CCH * sizeof(float),
                   hipMemcpyDeviceToDevice, stream);
    hipMemcpyAsync(ebuf, e_in, (size_t)E * CCH * sizeof(float),
                   hipMemcpyDeviceToDevice, stream);

    // workspace: agg (N*C f32) then bf16 weight fragments
    float*  agg = (float*)d_ws;
    __bf16* wb  = (__bf16*)((char*)d_ws + (size_t)N * CCH * sizeof(float));
    // per-layer bf16 fragment counts
    const int EW0F = 6 * 4 * 512, EW1F = 2 * 4 * 512, EW2F = 2 * 4 * 512;
    const int NW0F = 4 * 4 * 512, NW1F = 2 * 4 * 512, NW2F = 2 * 4 * 512;
    const int PERL = EW0F + EW1F + EW2F + NW0F + NW1F + NW2F;  // 36864

    for (int l = 0; l < LAYERS; ++l) {
        __bf16* lw = wb + (size_t)l * PERL;
        swizzle_weight_kernel<<<6 * 4, 32, 0, stream>>>(ew0 + (size_t)l * 192 * 64, lw);
        swizzle_weight_kernel<<<2 * 4, 32, 0, stream>>>(ew1 + (size_t)l * 64 * 64,  lw + EW0F);
        swizzle_weight_kernel<<<2 * 4, 32, 0, stream>>>(ew2 + (size_t)l * 64 * 64,  lw + EW0F + EW1F);
        swizzle_weight_kernel<<<4 * 4, 32, 0, stream>>>(nw0 + (size_t)l * 128 * 64, lw + EW0F + EW1F + EW2F);
        swizzle_weight_kernel<<<2 * 4, 32, 0, stream>>>(nw1 + (size_t)l * 64 * 64,  lw + EW0F + EW1F + EW2F + NW0F);
        swizzle_weight_kernel<<<2 * 4, 32, 0, stream>>>(nw2 + (size_t)l * 64 * 64,  lw + EW0F + EW1F + EW2F + NW0F + NW1F);
    }

    const int edgeBlocks = (E + 63) / 64;
    const int nodeBlocks = (N + 63) / 64;

    for (int l = 0; l < LAYERS; ++l) {
        __bf16* lw = wb + (size_t)l * PERL;
        hipMemsetAsync(agg, 0, (size_t)N * CCH * sizeof(float), stream);
        edge_mlp_kernel<<<edgeBlocks, 128, 64 * 1024, stream>>>(
            xbuf, ebuf, srcI, dstI,
            lw, lw + EW0F, lw + EW0F + EW1F,
            eb0 + (size_t)l * CCH, eb1 + (size_t)l * CCH, eb2 + (size_t)l * CCH,
            eg + (size_t)l * CCH, ebt + (size_t)l * CCH,
            agg, E);
        node_mlp_kernel<<<nodeBlocks, 128, 56 * 1024, stream>>>(
            xbuf, agg,
            lw + EW0F + EW1F + EW2F,
            lw + EW0F + EW1F + EW2F + NW0F,
            lw + EW0F + EW1F + EW2F + NW0F + NW1F,
            nb0 + (size_t)l * CCH, nb1 + (size_t)l * CCH, nb2 + (size_t)l * CCH,
            ng + (size_t)l * CCH, nbt + (size_t)l * CCH,
            N);
    }
}
